// AFDChannelAttention_429496730011
// MI455X (gfx1250) — matrix operand order repaired and checked by reading, not compiled, not run on hardware
//
#include <hip/hip_runtime.h>

typedef __bf16 bf16;
typedef __attribute__((ext_vector_type(16))) bf16  v16bf;
typedef __attribute__((ext_vector_type(8)))  float v8f;

#define LTOT 4096
#define NB   64
#define ROWS_TOTAL (32 * 512)   // b*C = 16384 rows of length L
#define TILE_M 16
#define SPLIT 4                 // waves cooperating on one row-tile (split over L)
#define TILES_PER_BLOCK 2
#define THREADS (SPLIT * TILES_PER_BLOCK * 32)  // 256 threads = 8 waves

// ---------------------------------------------------------------------------
// Setup: Blaschke basis in two bf16 layouts.
//   Bb[n][l]  (row-major over l)  -> phase-1 B operand (K = l contiguous)
//   Bt[l][n]  (row-major over n)  -> phase-3 B operand (K = n contiguous)
// ---------------------------------------------------------------------------
__global__ void afd_basis_kernel(bf16* __restrict__ Bb, bf16* __restrict__ Bt) {
    int idx = blockIdx.x * blockDim.x + threadIdx.x;   // 0 .. 64*4096-1
    int n = idx >> 12;
    int l = idx & (LTOT - 1);
    const float TWOPI = 6.283185307179586f;
    float theta = TWOPI * (float)n * (1.0f / 64.0f);
    float t     = TWOPI * (float)l * (1.0f / (float)(LTOT - 1)); // endpoint included
    float cosp  = __cosf(t - theta);
    // sqrt(1-r^2) * (1 - r cos) / (1 - 2 r cos + r^2), r = 0.9
    float val = 0.43588989435406736f * (1.0f - 0.9f * cosp) /
                (1.0f - 1.8f * cosp + 0.81f);
    bf16 v = (bf16)val;
    Bb[n * LTOT + l] = v;
    Bt[l * NB + n]   = v;
}

// ---------------------------------------------------------------------------
// Fused proj -> softmax -> attn + residual.
// Each wave handles one 16-row tile jointly with SPLIT-1 sibling waves,
// splitting the L dimension both for the projection and the reconstruction.
// ---------------------------------------------------------------------------
__global__ __launch_bounds__(THREADS)
void afd_attn_kernel(const float* __restrict__ x,
                     const bf16*  __restrict__ Bb,
                     const bf16*  __restrict__ Bt,
                     float*       __restrict__ out) {
    __shared__ float sproj[TILES_PER_BLOCK][TILE_M][NB];
    __shared__ float swgt [TILES_PER_BLOCK][TILE_M][NB];

    const int tid  = threadIdx.x;
    const int lane = tid & 31;
    const int wave = tid >> 5;
    const int rt   = wave / SPLIT;        // row-tile within block (0..1)
    const int ws   = wave & (SPLIT - 1);  // L-split index (0..3)
    const int half = lane >> 4;           // 0: lanes 0-15, 1: lanes 16-31
    const int l16  = lane & 15;

    // Zero LDS projection accumulators.
    for (int i = tid; i < TILES_PER_BLOCK * TILE_M * NB; i += THREADS)
        ((float*)sproj)[i] = 0.0f;
    __syncthreads();

    const long tile = (long)blockIdx.x * TILES_PER_BLOCK + rt;
    const long base = tile * (long)TILE_M * LTOT;

    // ------------------- Phase 1: proj = X * B^T -------------------
    v8f acc0 = {}, acc1 = {}, acc2 = {}, acc3 = {};
    {
        const float* xrow  = x + base + (long)l16 * LTOT;  // row M = l16
        const int kstart = ws * (LTOT / SPLIT);
        const int kend   = kstart + (LTOT / SPLIT);
        for (int kk = kstart; kk < kend; kk += 32) {
            // A operand, contiguous k order used here (B matches): lanes 0-15 hold K = {0..15},
            //                  lanes 16-31 hold K = {16..31}
            const int o1 = kk + half * 16;
            const int o2 = kk + half * 16 + 8;
            float4 f0 = *(const float4*)(xrow + o1);
            float4 f1 = *(const float4*)(xrow + o1 + 4);
            float4 f2 = *(const float4*)(xrow + o2);
            float4 f3 = *(const float4*)(xrow + o2 + 4);
            v16bf a;
            a[0]  = (bf16)f0.x; a[1]  = (bf16)f0.y; a[2]  = (bf16)f0.z; a[3]  = (bf16)f0.w;
            a[4]  = (bf16)f1.x; a[5]  = (bf16)f1.y; a[6]  = (bf16)f1.z; a[7]  = (bf16)f1.w;
            a[8]  = (bf16)f2.x; a[9]  = (bf16)f2.y; a[10] = (bf16)f2.z; a[11] = (bf16)f2.w;
            a[12] = (bf16)f3.x; a[13] = (bf16)f3.y; a[14] = (bf16)f3.z; a[15] = (bf16)f3.w;
            // B layout: lane column N = l16 (basis n), K range = half*16 .. +15,
            // contiguous along l in Bb -> one 32B load per n-tile.
            const bf16* bp = Bb + (long)l16 * LTOT + (kk + half * 16);
            v16bf b0 = *(const v16bf*)(bp + 0L * 16 * LTOT);
            v16bf b1 = *(const v16bf*)(bp + 1L * 16 * LTOT);
            v16bf b2 = *(const v16bf*)(bp + 2L * 16 * LTOT);
            v16bf b3 = *(const v16bf*)(bp + 3L * 16 * LTOT);
            acc0 = __builtin_amdgcn_wmma_f32_16x16x32_bf16(false, a, false, b0, (short)0, acc0, false, false);
            acc1 = __builtin_amdgcn_wmma_f32_16x16x32_bf16(false, a, false, b1, (short)0, acc1, false, false);
            acc2 = __builtin_amdgcn_wmma_f32_16x16x32_bf16(false, a, false, b2, (short)0, acc2, false, false);
            acc3 = __builtin_amdgcn_wmma_f32_16x16x32_bf16(false, a, false, b3, (short)0, acc3, false, false);
        }
    }
    // Merge the SPLIT partial projections: D layout lane -> (M = half*8+r, N = l16).
    #pragma unroll
    for (int r = 0; r < 8; ++r) {
        const int m = half * 8 + r;
        atomicAdd(&sproj[rt][m][0 * 16 + l16], acc0[r]);
        atomicAdd(&sproj[rt][m][1 * 16 + l16], acc1[r]);
        atomicAdd(&sproj[rt][m][2 * 16 + l16], acc2[r]);
        atomicAdd(&sproj[rt][m][3 * 16 + l16], acc3[r]);
    }
    __syncthreads();

    // ------------------- Phase 2: softmax(|proj|) per row -------------------
    if (ws == 0 && lane < TILE_M) {
        float mx = 0.0f;
        #pragma unroll 8
        for (int n = 0; n < NB; ++n)
            mx = fmaxf(mx, fabsf(sproj[rt][lane][n]));
        float sum = 0.0f;
        #pragma unroll 8
        for (int n = 0; n < NB; ++n) {
            float e = __expf(fabsf(sproj[rt][lane][n]) - mx);
            swgt[rt][lane][n] = e;
            sum += e;
        }
        float inv = 1.0f / sum;
        #pragma unroll 8
        for (int n = 0; n < NB; ++n)
            swgt[rt][lane][n] *= inv;
    }
    __syncthreads();

    // ------------------- Phase 3: out = x + W * B -------------------
    // A operand (weights, 16 x 64 = two K=32 steps) loaded once from LDS.
    v16bf aw0, aw1;
    {
        const float* wrow = &swgt[rt][l16][0];   // row M = l16
        #pragma unroll
        for (int s = 0; s < 2; ++s) {
            const int o1 = s * 32 + half * 16;
            const int o2 = s * 32 + half * 16 + 8;
            float4 f0 = *(const float4*)(wrow + o1);
            float4 f1 = *(const float4*)(wrow + o1 + 4);
            float4 f2 = *(const float4*)(wrow + o2);
            float4 f3 = *(const float4*)(wrow + o2 + 4);
            v16bf a;
            a[0]  = (bf16)f0.x; a[1]  = (bf16)f0.y; a[2]  = (bf16)f0.z; a[3]  = (bf16)f0.w;
            a[4]  = (bf16)f1.x; a[5]  = (bf16)f1.y; a[6]  = (bf16)f1.z; a[7]  = (bf16)f1.w;
            a[8]  = (bf16)f2.x; a[9]  = (bf16)f2.y; a[10] = (bf16)f2.z; a[11] = (bf16)f2.w;
            a[12] = (bf16)f3.x; a[13] = (bf16)f3.y; a[14] = (bf16)f3.z; a[15] = (bf16)f3.w;
            if (s == 0) aw0 = a; else aw1 = a;
        }
    }
    {
        const int lstart = ws * (LTOT / SPLIT);
        const int lend   = lstart + (LTOT / SPLIT);
        const int mbase  = half * 8;
        for (int l0 = lstart; l0 < lend; l0 += 16) {
            const int col = l0 + l16;     // output column l for this lane
            // Residual fused into the accumulator: C preloaded with x.
            v8f c;
            #pragma unroll
            for (int r = 0; r < 8; ++r)
                c[r] = x[base + (long)(mbase + r) * LTOT + col];
            // B operand: K = n contiguous in Bt[l][n]; lane K-range = s*32 + half*16.
            const bf16* btp = Bt + (long)col * NB + half * 16;
            v16bf b0 = *(const v16bf*)(btp);
            v16bf b1 = *(const v16bf*)(btp + 32);
            c = __builtin_amdgcn_wmma_f32_16x16x32_bf16(false, aw0, false, b0, (short)0, c, false, false);
            c = __builtin_amdgcn_wmma_f32_16x16x32_bf16(false, aw1, false, b1, (short)0, c, false, false);
            #pragma unroll
            for (int r = 0; r < 8; ++r)
                out[base + (long)(mbase + r) * LTOT + col] = c[r];
        }
    }
}

extern "C" void kernel_launch(void* const* d_in, const int* in_sizes, int n_in,
                              void* d_out, int out_size, void* d_ws, size_t ws_size,
                              hipStream_t stream) {
    (void)in_sizes; (void)n_in; (void)out_size; (void)ws_size;
    const float* x = (const float*)d_in[0];
    float* out = (float*)d_out;

    bf16* Bb = (bf16*)d_ws;                                      // 64*4096*2 = 512 KB
    bf16* Bt = (bf16*)((char*)d_ws + (size_t)NB * LTOT * sizeof(bf16)); // +512 KB

    // Basis: 64*4096 = 262144 elements.
    afd_basis_kernel<<<(NB * LTOT) / 256, 256, 0, stream>>>(Bb, Bt);

    // 16384 rows / (16 rows * 2 tiles per block) = 512 blocks of 8 waves.
    const int blocks = ROWS_TOTAL / (TILE_M * TILES_PER_BLOCK);
    afd_attn_kernel<<<blocks, THREADS, 0, stream>>>(x, Bb, Bt, out);
}
